// SDPA_29652454211669
// MI455X (gfx1250) — matrix operand order repaired and checked by reading, not compiled, not run on hardware
//
#include <hip/hip_runtime.h>
#include <stdint.h>

// ---------------------------------------------------------------------------
// (B,H,S,D) = (2,16,2048,64); softmax is over the HEAD axis (axis=1), so the
// full [B,H,S,S] score tensor must be materialized before normalization.
// ---------------------------------------------------------------------------
#define BATCH 2
#define HEADS 16
#define SEQ   2048
#define HDIM  64
#define SCALE 0.125f   // 1/sqrt(64)

typedef __attribute__((ext_vector_type(16))) __bf16 v16bf;
typedef __attribute__((ext_vector_type(8)))  __bf16 v8bf;
typedef __attribute__((ext_vector_type(8)))  float  v8f;

union Frag16 { v16bf v; v8bf h[2]; };

__device__ __forceinline__ unsigned short f32_to_bf16(float f) {
    unsigned int u = __float_as_uint(f);
    unsigned int r = (u + 0x7FFFu + ((u >> 16) & 1u)) >> 16;
    return (unsigned short)r;
}

// A-matrix fragment, 16x32 bf16 (M x K), source row-major, leading dim ld.
// lanes 0-15: row=lane, K {0..7,16..23}; lanes 16-31: row=lane-16, K {8..15,24..31}.
__device__ __forceinline__ v16bf load_frag_a(const unsigned short* base, int ld, int lane) {
    int row  = lane & 15;
    int koff = (lane >> 4) << 3;           // 0 or 8
    const unsigned short* p = base + (size_t)row * ld + koff;
    Frag16 f;
    f.h[0] = *reinterpret_cast<const v8bf*>(p);
    f.h[1] = *reinterpret_cast<const v8bf*>(p + 16);
    return f.v;
}

// B-matrix fragment, 32x16 bf16 (K x N), fed from "N-major, K contiguous"
// storage (K^T tile / pre-transposed V), leading dim ld.
// lanes 0-15: col=lane, K {0..7,16..23}; lanes 16-31: col=lane-16, K {8..15,24..31}.
__device__ __forceinline__ v16bf load_frag_b(const unsigned short* base, int ld, int lane) {
    int col  = lane & 15;
    int koff = (lane >> 4) << 3;           // 0 or 8
    const unsigned short* p = base + (size_t)col * ld + koff;
    Frag16 f;
    f.h[0] = *reinterpret_cast<const v8bf*>(p);
    f.h[1] = *reinterpret_cast<const v8bf*>(p + 16);
    return f.v;
}

// ---------------------------------------------------------------------------
// Phase 0: fp32 -> bf16 for Q,K; V converted AND transposed to [bh, d, s].
// ---------------------------------------------------------------------------
__global__ void cvt_kernel(const float* __restrict__ q, const float* __restrict__ k,
                           const float* __restrict__ v,
                           unsigned short* __restrict__ Qb,
                           unsigned short* __restrict__ Kb,
                           unsigned short* __restrict__ VT) {
    size_t i = (size_t)blockIdx.x * blockDim.x + threadIdx.x;
    const size_t total = (size_t)BATCH * HEADS * SEQ * HDIM;
    if (i >= total) return;
    Qb[i] = f32_to_bf16(q[i]);
    Kb[i] = f32_to_bf16(k[i]);
    size_t bh  = i / ((size_t)SEQ * HDIM);
    size_t rem = i % ((size_t)SEQ * HDIM);
    size_t s = rem / HDIM, d = rem % HDIM;
    VT[bh * (size_t)HDIM * SEQ + d * SEQ + s] = f32_to_bf16(v[i]);
}

// ---------------------------------------------------------------------------
// Phase 1: scores = (Q K^T)*scale, masked.  One wave owns a 16x64 output
// block (4 accumulators): the Q fragment is loaded once per 32-deep k-step
// and reused across 4 WMMAs.  Scores are streamed out non-temporally (536 MB
// write-once tensor; keep it out of the 192 MB L2).
// ---------------------------------------------------------------------------
__global__ void qk_kernel(const unsigned short* __restrict__ Qb,
                          const unsigned short* __restrict__ Kb,
                          const int* __restrict__ mask,
                          float* __restrict__ scores) {
    const int lane = threadIdx.x & 31;
    const int wave = threadIdx.x >> 5;
    const int qt = blockIdx.x;                 // 0..127 (q tile of 16)
    const int kb = blockIdx.y * 8 + wave;      // 0..31 (k block of 64)
    const int bh = blockIdx.z;                 // 0..31
    const int b  = bh / HEADS;

    const unsigned short* Qtile = Qb + ((size_t)bh * SEQ + qt * 16) * HDIM;
    const unsigned short* Kbase = Kb + ((size_t)bh * SEQ + kb * 64) * HDIM;

    v8f acc[4] = {};
#pragma unroll
    for (int kk = 0; kk < HDIM; kk += 32) {
        v16bf a = load_frag_a(Qtile + kk, HDIM, lane);
#pragma unroll
        for (int j = 0; j < 4; ++j) {
            v16bf bb = load_frag_b(Kbase + (size_t)(j * 16) * HDIM + kk, HDIM, lane);
            acc[j] = __builtin_amdgcn_wmma_f32_16x16x32_bf16(false, a, false, bb,
                                                             (short)0, acc[j], false, false);
        }
    }

    const int n  = lane & 15;
    const int rb = (lane >> 4) << 3;           // 0 or 8
    const int* mrow = mask + ((size_t)b * SEQ + qt * 16) * SEQ;
    float* srow = scores + ((size_t)bh * SEQ + qt * 16) * SEQ;
#pragma unroll
    for (int j = 0; j < 4; ++j) {
        int col = kb * 64 + j * 16 + n;
#pragma unroll
        for (int r = 0; r < 8; ++r) {
            int row = rb + r;
            float sv = acc[j][r] * SCALE;
            int m = mrow[(size_t)row * SEQ + col];
            __builtin_nontemporal_store((m == 0) ? -1e9f : sv,
                                        &srow[(size_t)row * SEQ + col]);
        }
    }
}

// ---------------------------------------------------------------------------
// Phase 2: softmax across the 16 HEAD planes for each (b,q,k); reads scores
// non-temporally (read-once stream), writes bf16 attention weights.
// ---------------------------------------------------------------------------
__global__ void softmax_h_kernel(const float* __restrict__ scores,
                                 unsigned short* __restrict__ attn) {
    size_t idx = (size_t)blockIdx.x * blockDim.x + threadIdx.x;
    const size_t per_b = (size_t)SEQ * SEQ;
    if (idx >= (size_t)BATCH * per_b) return;
    size_t b  = idx / per_b;
    size_t qk = idx - b * per_b;
    size_t base = b * HEADS * per_b + qk;

    float s[HEADS];
    float mx = -3.0e38f;
#pragma unroll
    for (int h = 0; h < HEADS; ++h) {
        s[h] = __builtin_nontemporal_load(&scores[base + (size_t)h * per_b]);
        mx = fmaxf(mx, s[h]);
    }
    float sum = 0.f;
#pragma unroll
    for (int h = 0; h < HEADS; ++h) {
        s[h] = __expf(s[h] - mx);
        sum += s[h];
    }
    float inv = 1.0f / sum;
#pragma unroll
    for (int h = 0; h < HEADS; ++h)
        attn[base + (size_t)h * per_b] = f32_to_bf16(s[h] * inv);
}

// ---------------------------------------------------------------------------
// Phase 3: out = attn @ V.  One wave owns a full 16x64 output block (all of
// D): per 32-deep k-step it loads its unique attn fragment ONCE and runs 4
// WMMAs against the 4 shared V^T fragments (which hit L0/L2 across the 8
// waves of the block).  attn is therefore read from HBM exactly once.
// ---------------------------------------------------------------------------
__global__ void av_kernel(const unsigned short* __restrict__ attn,
                          const unsigned short* __restrict__ VT,
                          float* __restrict__ out) {
    const int lane = threadIdx.x & 31;
    const int wave = threadIdx.x >> 5;
    const int qt = blockIdx.x * 8 + wave;      // 0..127 (q tile of 16)
    const int bh = blockIdx.y;                 // 0..31

    const unsigned short* Atile = attn + ((size_t)bh * SEQ + qt * 16) * SEQ;
    const unsigned short* Vbase = VT + (size_t)bh * HDIM * SEQ;

    v8f acc[4] = {};
    for (int kk = 0; kk < SEQ; kk += 32) {
        __builtin_prefetch(Atile + kk + 128, 0, 0);
        v16bf a = load_frag_a(Atile + kk, SEQ, lane);
#pragma unroll
        for (int j = 0; j < 4; ++j) {
            v16bf bb = load_frag_b(Vbase + (size_t)(j * 16) * SEQ + kk, SEQ, lane);
            acc[j] = __builtin_amdgcn_wmma_f32_16x16x32_bf16(false, a, false, bb,
                                                             (short)0, acc[j], false, false);
        }
    }

    const int n  = lane & 15;
    const int rb = (lane >> 4) << 3;
    float* orow = out + ((size_t)bh * SEQ + qt * 16) * HDIM;
#pragma unroll
    for (int j = 0; j < 4; ++j)
#pragma unroll
        for (int r = 0; r < 8; ++r)
            orow[(size_t)(rb + r) * HDIM + j * 16 + n] = acc[j][r];
}

// ---------------------------------------------------------------------------
// Host-side launch.  Workspace layout (bytes):
//   Qb  bf16 [B,H,S,D] :        0 ..   8388608
//   Kb  bf16 [B,H,S,D] :  8388608 ..  16777216
//   VT  bf16 [B,H,D,S] : 16777216 ..  25165824
//   scores f32 [B,H,S,S]: 25165824 .. 562036736
//   attn  bf16 [B,H,S,S]: 562036736 .. 830472192   (~792 MiB total)
// ---------------------------------------------------------------------------
extern "C" void kernel_launch(void* const* d_in, const int* in_sizes, int n_in,
                              void* d_out, int out_size, void* d_ws, size_t ws_size,
                              hipStream_t stream) {
    (void)in_sizes; (void)n_in; (void)out_size; (void)ws_size;
    const float* q = (const float*)d_in[0];
    const float* k = (const float*)d_in[1];
    const float* v = (const float*)d_in[2];
    const int*   mask = (const int*)d_in[3];
    float* out = (float*)d_out;

    char* ws = (char*)d_ws;
    const size_t nqkv = (size_t)BATCH * HEADS * SEQ * HDIM;     // 4,194,304
    const size_t nss  = (size_t)BATCH * HEADS * SEQ * SEQ;      // 134,217,728
    unsigned short* Qb = (unsigned short*)(ws);
    unsigned short* Kb = (unsigned short*)(ws + nqkv * 2);
    unsigned short* VT = (unsigned short*)(ws + nqkv * 4);
    float*      scores = (float*)(ws + nqkv * 6);
    unsigned short* attn = (unsigned short*)(ws + nqkv * 6 + nss * 4);

    // Phase 0: convert + transpose V
    cvt_kernel<<<dim3((unsigned)(nqkv / 256)), 256, 0, stream>>>(q, k, v, Qb, Kb, VT);

    // Phase 1: QK^T (scale + mask fused); each wave -> 16x64 score block
    qk_kernel<<<dim3(SEQ / 16, (SEQ / 64) / 8, BATCH * HEADS), 256, 0, stream>>>(
        Qb, Kb, mask, scores);

    // Phase 2: softmax over heads
    const size_t nbqk = (size_t)BATCH * SEQ * SEQ;              // 8,388,608
    softmax_h_kernel<<<dim3((unsigned)(nbqk / 256)), 256, 0, stream>>>(scores, attn);

    // Phase 3: attn @ V; each wave -> 16x64 (full-D) output block
    av_kernel<<<dim3((SEQ / 16) / 8, BATCH * HEADS), 256, 0, stream>>>(attn, VT, out);
}
